// RWKV_30631706755511
// MI455X (gfx1250) — matrix operand order repaired and checked by reading, not compiled, not run on hardware
//
#include <hip/hip_runtime.h>

// ---------------------------------------------------------------------------
// RWKV v4 block on gfx1250 (MI455X).
// GEMMs: bf16 WMMA (v_wmma_f32_16x16x32_bf16), f32 accumulate, fused epilogues.
//   A tile staged in LDS (async-copy path when the toolchain exposes the
//   gfx1250 global_load_async_to_lds builtins, plain ds_store fallback),
//   B fragments streamed from global/L2 (weights are L2-resident: 26 MB bf16
//   vs 192 MB L2).
// WKV scan: 1 thread per (b,d) channel, serial over L.
// ---------------------------------------------------------------------------

typedef __bf16 bf16_t;
typedef __attribute__((ext_vector_type(16))) __bf16 v16bf;
typedef __attribute__((ext_vector_type(8)))  __bf16 v8bf;
typedef __attribute__((ext_vector_type(8)))  float  v8f;

#if __has_builtin(__builtin_amdgcn_global_load_async_to_lds_b128) && \
    __has_builtin(__builtin_amdgcn_s_wait_asynccnt)
#define USE_ASYNC_LDS 1
// Builtin signature (from clang diagnostic): params are pointers to
// 'int __attribute__((vector_size(16)))' in AS1 (__device__) / AS3 (__shared__).
typedef int v4i_vs __attribute__((vector_size(16)));
typedef __attribute__((address_space(1))) v4i_vs as1_v4i;
typedef __attribute__((address_space(3))) v4i_vs as3_v4i;
#else
#define USE_ASYNC_LDS 0
#endif

__device__ __forceinline__ v8f zero_v8f() {
    v8f z;
#pragma unroll
    for (int i = 0; i < 8; ++i) z[i] = 0.0f;
    return z;
}

__device__ __forceinline__ float sigmoidf_dev(float x) {
    return 1.0f / (1.0f + expf(-x));
}

// ---------------------------------------------------------------------------
// Generic NT GEMM: C[m,n] = sum_k A[m,k] * Bw[n,k], A:(M,K) bf16, Bw:(N,K) bf16.
// Block tile 128(M) x 256(N), 8 waves (2 M x 4 N), wave tile 64x64 = 4x4 WMMA.
// A tile (128x32 bf16) double-buffered in LDS (pitch 40 halfs = conflict-free
// ds_load_b128); copied with async-to-LDS ops when available.
// Epilogue modes:
//   0: outF = acc
//   1: outH = bf16(acc)
//   2: outH = bf16(sigmoid(acc))
//   3: outF = aux0 + acc                      (residual add)
//   4: outH = bf16(relu(acc)^2)               (FFN squared-relu)
//   5: outF = aux0 + float(auxs) * acc        (final: x1 + sig(r)*ffn_v)
// ---------------------------------------------------------------------------
#define GEMM_TM 4
#define GEMM_TN 4
#define LDS_PITCH 40   // halfs per staged A row (32 data + 8 pad)

__global__ __launch_bounds__(256) void gemm_nt_wmma(
    const bf16_t* __restrict__ A, const bf16_t* __restrict__ Bw,
    int M, int N, int K,
    float* __restrict__ outF, bf16_t* __restrict__ outH,
    const float* __restrict__ aux0, const bf16_t* __restrict__ auxs,
    int mode)
{
    __shared__ bf16_t smA[2][128 * LDS_PITCH];   // 2 x 10 KB

    const int tid  = threadIdx.x;
    const int lane = tid & 31;
    const int wave = tid >> 5;
    const int l16  = lane & 15;
    const int hi   = lane >> 4;        // half-wave select

    const int mBlk  = blockIdx.y * 128;
    const int mwLoc = (wave & 1) * 64;                   // wave M base (local)
    const int nw    = blockIdx.x * 256 + (wave >> 1) * 64;  // wave N base

    // A-staging assignment: 256 threads x 32 B = 8 KB tile.
    const int stRow = tid >> 1;          // 0..127
    const int stCol = (tid & 1) * 16;    // halfs
    const bf16_t* aSrcBase = A + (size_t)(mBlk + stRow) * (size_t)K + stCol;
    bf16_t* const stDst0 = &smA[0][stRow * LDS_PITCH + stCol];
    bf16_t* const stDst1 = &smA[1][stRow * LDS_PITCH + stCol];

    const size_t sK = (size_t)K;

    v8f acc[GEMM_TM][GEMM_TN];
#pragma unroll
    for (int i = 0; i < GEMM_TM; ++i)
#pragma unroll
        for (int j = 0; j < GEMM_TN; ++j)
            acc[i][j] = zero_v8f();

    auto stageA = [&](bf16_t* dst, int kt) {
        const bf16_t* src = aSrcBase + kt;
#if USE_ASYNC_LDS
        as1_v4i* gs = (as1_v4i*)(unsigned long long)src;
        as3_v4i* ld = (as3_v4i*)(unsigned long long)dst;
        __builtin_amdgcn_global_load_async_to_lds_b128(gs,     ld,     0, 0);
        __builtin_amdgcn_global_load_async_to_lds_b128(gs + 1, ld + 1, 0, 0);
#else
        *reinterpret_cast<v8bf*>(dst)     = *reinterpret_cast<const v8bf*>(src);
        *reinterpret_cast<v8bf*>(dst + 8) = *reinterpret_cast<const v8bf*>(src + 8);
#endif
    };

    stageA(stDst0, 0);

    for (int kt = 0; kt < K; kt += 32) {
        const int cur = (kt >> 5) & 1;
        const bf16_t* smCur = smA[cur];

#if USE_ASYNC_LDS
        __builtin_amdgcn_s_wait_asynccnt(0);  // cur buffer landed in LDS
#endif
        __syncthreads();                      // visible to all; prior readers done
        if (kt + 32 < K) {
            stageA(cur ? stDst0 : stDst1, kt + 32);
            __builtin_prefetch(Bw + (size_t)(nw + l16) * sK + kt + 32, 0, 3);
        }

        // B fragments (32x16): lane holds 16 contiguous K at base hi*16, col l16
        v16bf bfr[GEMM_TN];
#pragma unroll
        for (int j = 0; j < GEMM_TN; ++j) {
            const bf16_t* pb = Bw + (size_t)(nw + j * 16 + l16) * sK + kt + hi * 16;
            bfr[j] = *reinterpret_cast<const v16bf*>(pb);
        }

        // A fragments from LDS: lane hi=0: K[0..15]; hi=1: K[16..31]
        union { v16bf v; v8bf h[2]; } afr[GEMM_TM];
#pragma unroll
        for (int i = 0; i < GEMM_TM; ++i) {
            const bf16_t* pa = smCur + (mwLoc + i * 16 + l16) * LDS_PITCH + hi * 16;
            afr[i].h[0] = *reinterpret_cast<const v8bf*>(pa);
            afr[i].h[1] = *reinterpret_cast<const v8bf*>(pa + 8);
        }

#pragma unroll
        for (int i = 0; i < GEMM_TM; ++i)
#pragma unroll
            for (int j = 0; j < GEMM_TN; ++j)
                acc[i][j] = __builtin_amdgcn_wmma_f32_16x16x32_bf16(
                    false, afr[i].v, false, bfr[j],
                    (short)0, acc[i][j], false, false);

        __syncthreads();                      // done reading cur before overwrite
    }

    // Epilogue. C layout: VGPR e holds M = e + 8*hi, N = l16 (16x16 f32 tile).
    const int mw = mBlk + mwLoc;
#pragma unroll
    for (int i = 0; i < GEMM_TM; ++i) {
#pragma unroll
        for (int j = 0; j < GEMM_TN; ++j) {
            const int mBase = mw + i * 16 + hi * 8;
            const int n     = nw + j * 16 + l16;
#pragma unroll
            for (int e = 0; e < 8; ++e) {
                const size_t idx = (size_t)(mBase + e) * (size_t)N + n;
                const float c = acc[i][j][e];
                if (mode == 0) {
                    outF[idx] = c;
                } else if (mode == 1) {
                    outH[idx] = (bf16_t)c;
                } else if (mode == 2) {
                    outH[idx] = (bf16_t)sigmoidf_dev(c);
                } else if (mode == 3) {
                    outF[idx] = aux0[idx] + c;
                } else if (mode == 4) {
                    const float t = fmaxf(c, 0.0f);
                    outH[idx] = (bf16_t)(t * t);
                } else { // 5
                    outF[idx] = aux0[idx] + (float)auxs[idx] * c;
                }
            }
        }
    }
}

// ---------------------------------------------------------------------------
// LayerNorm: one block (256 threads) per token, D = 1024 (4 floats/thread).
// ---------------------------------------------------------------------------
__global__ __launch_bounds__(256) void ln_kernel(
    const float* __restrict__ x, const float* __restrict__ w,
    const float* __restrict__ b, float* __restrict__ y, int D)
{
    const size_t row = blockIdx.x;
    const int d0 = threadIdx.x * 4;
    const float4 v = *reinterpret_cast<const float4*>(x + row * (size_t)D + d0);

    float s  = v.x + v.y + v.z + v.w;
    float s2 = v.x * v.x + v.y * v.y + v.z * v.z + v.w * v.w;
#pragma unroll
    for (int off = 16; off > 0; off >>= 1) {
        s  += __shfl_xor(s,  off, 32);
        s2 += __shfl_xor(s2, off, 32);
    }
    __shared__ float sm1[8], sm2[8];
    const int wv = threadIdx.x >> 5, ln = threadIdx.x & 31;
    if (ln == 0) { sm1[wv] = s; sm2[wv] = s2; }
    __syncthreads();
    if (threadIdx.x == 0) {
        float t1 = 0.0f, t2 = 0.0f;
        for (int i = 0; i < 8; ++i) { t1 += sm1[i]; t2 += sm2[i]; }
        sm1[0] = t1; sm2[0] = t2;
    }
    __syncthreads();
    const float mean = sm1[0] / (float)D;
    const float var  = sm2[0] / (float)D - mean * mean;
    const float rstd = rsqrtf(var + 1e-5f);

    float4 o;
    o.x = (v.x - mean) * rstd * w[d0 + 0] + b[d0 + 0];
    o.y = (v.y - mean) * rstd * w[d0 + 1] + b[d0 + 1];
    o.z = (v.z - mean) * rstd * w[d0 + 2] + b[d0 + 2];
    o.w = (v.w - mean) * rstd * w[d0 + 3] + b[d0 + 3];
    *reinterpret_cast<float4*>(y + row * (size_t)D + d0) = o;
}

// ---------------------------------------------------------------------------
// Time-shift mixing: out = cur*mix + prev*(1-mix), prev = token l-1 (0 at l=0).
// ---------------------------------------------------------------------------
__global__ void mix3_kernel(
    const float* __restrict__ xa,
    const float* __restrict__ mk, const float* __restrict__ mv,
    const float* __restrict__ mr,
    bf16_t* __restrict__ ok, bf16_t* __restrict__ ov, bf16_t* __restrict__ orr,
    int L, int D, size_t total)
{
    const size_t i = (size_t)blockIdx.x * blockDim.x + threadIdx.x;
    if (i >= total) return;
    const size_t m = i / D;
    const int d = (int)(i - m * D);
    const int l = (int)(m % L);
    const float cur  = xa[i];
    const float prev = (l == 0) ? 0.0f : xa[i - D];
    const float fk = mk[d], fv = mv[d], fr = mr[d];
    ok[i]  = (bf16_t)(cur * fk + prev * (1.0f - fk));
    ov[i]  = (bf16_t)(cur * fv + prev * (1.0f - fv));
    orr[i] = (bf16_t)(cur * fr + prev * (1.0f - fr));
}

__global__ void mix2_kernel(
    const float* __restrict__ xa,
    const float* __restrict__ mk, const float* __restrict__ mr,
    bf16_t* __restrict__ ok, bf16_t* __restrict__ orr,
    int L, int D, size_t total)
{
    const size_t i = (size_t)blockIdx.x * blockDim.x + threadIdx.x;
    if (i >= total) return;
    const size_t m = i / D;
    const int d = (int)(i - m * D);
    const int l = (int)(m % L);
    const float cur  = xa[i];
    const float prev = (l == 0) ? 0.0f : xa[i - D];
    const float fk = mk[d], fr = mr[d];
    ok[i]  = (bf16_t)(cur * fk + prev * (1.0f - fk));
    orr[i] = (bf16_t)(cur * fr + prev * (1.0f - fr));
}

// ---------------------------------------------------------------------------
// Numerically-stable WKV scan; fuses s = sigmoid(r) * wkv -> bf16.
// One thread per (b,d) channel; lanes stride d => coalesced per step.
// ---------------------------------------------------------------------------
__global__ void wkv_kernel(
    const float* __restrict__ kF, const float* __restrict__ vF,
    const bf16_t* __restrict__ rsig,
    const float* __restrict__ tdec, const float* __restrict__ tfirst,
    bf16_t* __restrict__ sOut, int Bn, int L, int D)
{
    const int ch = blockIdx.x * blockDim.x + threadIdx.x;
    if (ch >= Bn * D) return;
    const int b = ch / D;
    const int d = ch - b * D;
    const float w = -expf(tdec[d]);
    const float u = tfirst[d];
    float aa = 0.0f, bb = 0.0f, pp = -1e38f;
    const size_t base = (size_t)b * L * D + d;
    for (int t = 0; t < L; ++t) {
        const size_t idx = base + (size_t)t * D;
        const float kk = kF[idx];
        const float vv = vF[idx];
        const float ww = u + kk;
        const float q  = fmaxf(pp, ww);
        const float e1 = expf(pp - q);
        const float e2 = expf(ww - q);
        const float out = (e1 * aa + e2 * vv) / (e1 * bb + e2);
        sOut[idx] = (bf16_t)((float)rsig[idx] * out);
        const float ww2 = pp + w;
        const float q2  = fmaxf(ww2, kk);
        const float e1b = expf(ww2 - q2);
        const float e2b = expf(kk - q2);
        aa = e1b * aa + e2b * vv;
        bb = e1b * bb + e2b;
        pp = q2;
    }
}

// ---------------------------------------------------------------------------
__global__ void cvt_bf16_kernel(const float* __restrict__ in,
                                bf16_t* __restrict__ out, size_t n)
{
    const size_t i = (size_t)blockIdx.x * blockDim.x + threadIdx.x;
    if (i < n) out[i] = (bf16_t)in[i];
}

// ---------------------------------------------------------------------------
extern "C" void kernel_launch(void* const* d_in, const int* in_sizes, int n_in,
                              void* d_out, int out_size, void* d_ws, size_t ws_size,
                              hipStream_t stream)
{
    constexpr int Bsz = 8, L = 2048, D = 1024, F = 4096;
    constexpr int M = Bsz * L;                 // 16384 tokens
    const size_t MD = (size_t)M * D;
    const size_t MF = (size_t)M * F;

    const float* x      = (const float*)d_in[0];
    const float* lnaw   = (const float*)d_in[1];
    const float* lnab   = (const float*)d_in[2];
    const float* lnfw   = (const float*)d_in[3];
    const float* lnfb   = (const float*)d_in[4];
    const float* tdec   = (const float*)d_in[5];
    const float* tfirst = (const float*)d_in[6];
    const float* amk    = (const float*)d_in[7];
    const float* amv    = (const float*)d_in[8];
    const float* amr    = (const float*)d_in[9];
    const float* Wk     = (const float*)d_in[10];
    const float* Wv     = (const float*)d_in[11];
    const float* Wr     = (const float*)d_in[12];
    const float* Wo     = (const float*)d_in[13];
    const float* fmk    = (const float*)d_in[14];
    const float* fmr    = (const float*)d_in[15];
    const float* fWk    = (const float*)d_in[16];
    const float* fWv    = (const float*)d_in[17];
    const float* fWr    = (const float*)d_in[18];

    char* wsp = (char*)d_ws;
    size_t off = 0;
    auto alloc = [&](size_t bytes) -> void* {
        void* p = wsp + off;
        off += (bytes + 255) & ~(size_t)255;
        return p;
    };

    // bf16 weight copies (all live in L2: 26 MB total)
    bf16_t* WkH  = (bf16_t*)alloc((size_t)D * D * 2);
    bf16_t* WvH  = (bf16_t*)alloc((size_t)D * D * 2);
    bf16_t* WrH  = (bf16_t*)alloc((size_t)D * D * 2);
    bf16_t* WoH  = (bf16_t*)alloc((size_t)D * D * 2);
    bf16_t* fWkH = (bf16_t*)alloc((size_t)F * D * 2);
    bf16_t* fWvH = (bf16_t*)alloc((size_t)D * F * 2);
    bf16_t* fWrH = (bf16_t*)alloc((size_t)D * D * 2);
    // activations
    float*  xa   = (float*) alloc(MD * 4);   // LN output (reused for both LNs)
    bf16_t* mkB  = (bf16_t*)alloc(MD * 2);   // mixed-k  (reused as kx)
    bf16_t* mvB  = (bf16_t*)alloc(MD * 2);   // mixed-v  (reused as rx)
    bf16_t* mrB  = (bf16_t*)alloc(MD * 2);   // mixed-r
    float*  kF   = (float*) alloc(MD * 4);
    float*  vF   = (float*) alloc(MD * 4);
    bf16_t* rsig = (bf16_t*)alloc(MD * 2);   // sigmoid(r)  (reused for ffn gate)
    bf16_t* sH   = (bf16_t*)alloc(MD * 2);   // sigmoid(r)*wkv
    float*  x1F  = (float*) alloc(MD * 4);   // x + att
    bf16_t* hH   = (bf16_t*)alloc(MF * 2);   // relu(kx@Wk)^2

    auto cvt = [&](const float* in, bf16_t* outp, size_t n) {
        cvt_bf16_kernel<<<(unsigned)((n + 255) / 256), 256, 0, stream>>>(in, outp, n);
    };
    auto gemm = [&](const bf16_t* A, const bf16_t* Bw, int Mm, int Nn, int Kk,
                    float* oF, bf16_t* oH, const float* a0, const bf16_t* as,
                    int mode) {
        dim3 grid(Nn / 256, Mm / 128);
        gemm_nt_wmma<<<grid, 256, 0, stream>>>(A, Bw, Mm, Nn, Kk, oF, oH, a0, as, mode);
    };

    // 0) weights -> bf16
    cvt(Wk,  WkH,  (size_t)D * D);
    cvt(Wv,  WvH,  (size_t)D * D);
    cvt(Wr,  WrH,  (size_t)D * D);
    cvt(Wo,  WoH,  (size_t)D * D);
    cvt(fWk, fWkH, (size_t)F * D);
    cvt(fWv, fWvH, (size_t)D * F);
    cvt(fWr, fWrH, (size_t)D * D);

    const unsigned mixGrid = (unsigned)((MD + 255) / 256);

    // 1) attention LN + time-shift mixes
    ln_kernel<<<M, 256, 0, stream>>>(x, lnaw, lnab, xa, D);
    mix3_kernel<<<mixGrid, 256, 0, stream>>>(xa, amk, amv, amr, mkB, mvB, mrB, L, D, MD);

    // 2) k/v/r projections (WMMA)
    gemm(mkB, WkH, M, D, D, kF, nullptr, nullptr, nullptr, 0);
    gemm(mvB, WvH, M, D, D, vF, nullptr, nullptr, nullptr, 0);
    gemm(mrB, WrH, M, D, D, nullptr, rsig, nullptr, nullptr, 2); // sigmoid fused

    // 3) WKV scan, fused with sigmoid(r) gating
    wkv_kernel<<<(Bsz * D + 255) / 256, 256, 0, stream>>>(
        kF, vF, rsig, tdec, tfirst, sH, Bsz, L, D);

    // 4) attention output projection + residual
    gemm(sH, WoH, M, D, D, x1F, nullptr, x, nullptr, 3);

    // 5) FFN LN + mixes
    ln_kernel<<<M, 256, 0, stream>>>(x1F, lnfw, lnfb, xa, D);
    mix2_kernel<<<mixGrid, 256, 0, stream>>>(xa, fmk, fmr, mkB, mvB, L, D, MD);

    // 6) FFN gate (sigmoid fused), FFN up (relu^2 fused)
    gemm(mvB, fWrH, M, D, D, nullptr, rsig, nullptr, nullptr, 2);
    gemm(mkB, fWkH, M, F, D, nullptr, hH, nullptr, nullptr, 4);

    // 7) FFN down + gate-multiply + residual -> final output (f32)
    gemm(hH, fWvH, M, D, F, (float*)d_out, nullptr, x1F, rsig, 5);
}
